// Block_9483287789913
// MI455X (gfx1250) — matrix operand order repaired and checked by reading, not compiled, not run on hardware
//
#include <hip/hip_runtime.h>
#include <cstdint>

// ---------------------------------------------------------------------------
// Transformer block (pre-norm MHA + FFN) for MI455X / gfx1250, bf16 WMMA.
// B=2, T=2048, E=1024, H=16, DH=64, FF=4096
// Compute-bound (~138 GFLOP vs ~6us of HBM traffic) => everything through
// v_wmma_f32_16x16x32_bf16; A-tiles staged via GLOBAL_LOAD_ASYNC_TO_LDS_B128
// with double-buffered LDS, B-tiles register-staged (transposed scatter).
// ---------------------------------------------------------------------------

typedef __attribute__((ext_vector_type(16))) __bf16 v16bf;
typedef __attribute__((ext_vector_type(8)))  __bf16 v8bf;
typedef __attribute__((ext_vector_type(8)))  float  v8f;
typedef __attribute__((ext_vector_type(8)))  unsigned short u16x8;

static constexpr int Bb  = 2;
static constexpr int Tt  = 2048;
static constexpr int Ee  = 1024;
static constexpr int Hh  = 16;
static constexpr int DHd = 64;
static constexpr int FFf = 4096;

#define GF_RELU  1
#define GF_BF16  2
#define GF_TRANS 4

__device__ __forceinline__ unsigned short f32_to_bf16(float f) {
    union { float f; unsigned u; } v; v.f = f;
    unsigned u = v.u;
    unsigned r = 0x7FFFu + ((u >> 16) & 1u);   // round-to-nearest-even
    return (unsigned short)((u + r) >> 16);
}

// ---- gfx1250 async copy to LDS (ASYNCcnt-tracked). LDS byte address is the
// low 32 bits of the generic pointer (flat LDS aperture keeps offset in
// ADDR[31:0], ISA 10.2). ----
__device__ __forceinline__ void async_load_b128(const unsigned short* gp,
                                                unsigned short* lp) {
    unsigned lds_addr = (unsigned)(uintptr_t)lp;
    unsigned long long ga = (unsigned long long)(uintptr_t)gp;
    asm volatile("global_load_async_to_lds_b128 %0, %1, off"
                 :: "v"(lds_addr), "v"(ga) : "memory");
}
__device__ __forceinline__ void wait_async0() {
    asm volatile("s_wait_asynccnt 0x0" ::: "memory");
}

// ---------------- elementwise fp32 -> bf16 convert -------------------------
__global__ __launch_bounds__(256) void cvt_f32_bf16_kernel(
    const float* __restrict__ in, unsigned short* __restrict__ out, int n) {
    int i = blockIdx.x * 256 + threadIdx.x;
    if (i < n) out[i] = f32_to_bf16(in[i]);
}

// ---------------- block reductions (256 threads, wave32) -------------------
__device__ __forceinline__ float block_sum(float v, float* red) {
    #pragma unroll
    for (int off = 16; off; off >>= 1) v += __shfl_xor(v, off, 32);
    __syncthreads();
    if ((threadIdx.x & 31) == 0) red[threadIdx.x >> 5] = v;
    __syncthreads();
    float s = 0.f;
    #pragma unroll
    for (int i = 0; i < 8; i++) s += red[i];
    return s;
}

__device__ __forceinline__ float block_max(float v, float* red) {
    #pragma unroll
    for (int off = 16; off; off >>= 1) v = fmaxf(v, __shfl_xor(v, off, 32));
    __syncthreads();
    if ((threadIdx.x & 31) == 0) red[threadIdx.x >> 5] = v;
    __syncthreads();
    float s = red[0];
    #pragma unroll
    for (int i = 1; i < 8; i++) s = fmaxf(s, red[i]);
    return s;
}

// ---------------- LayerNorm: one block (256 thr) per row of E --------------
__global__ __launch_bounds__(256) void ln_bf16_kernel(
    const float* __restrict__ x, const float* __restrict__ g,
    const float* __restrict__ b, unsigned short* __restrict__ out, int E) {
    __shared__ float red[8];
    const int row = blockIdx.x;
    const int tid = threadIdx.x;
    const float* xr = x + (size_t)row * E;
    unsigned short* orow = out + (size_t)row * E;

    float s = 0.f;
    for (int i = tid; i < E; i += 256) s += xr[i];
    float mean = block_sum(s, red) / (float)E;

    float vs = 0.f;
    for (int i = tid; i < E; i += 256) { float d = xr[i] - mean; vs += d * d; }
    float var = block_sum(vs, red) / (float)E;
    float inv = rsqrtf(var + 1e-5f);

    for (int i = tid; i < E; i += 256)
        orow[i] = f32_to_bf16((xr[i] - mean) * inv * g[i] + b[i]);
}

// ---------------- row softmax fp32 -> bf16 ---------------------------------
__global__ __launch_bounds__(256) void softmax_bf16_kernel(
    const float* __restrict__ S, unsigned short* __restrict__ P, int n) {
    __shared__ float red[8];
    const int row = blockIdx.x;
    const int tid = threadIdx.x;
    const float* s = S + (size_t)row * n;
    unsigned short* p = P + (size_t)row * n;

    float mx = -3.4e38f;
    for (int i = tid; i < n; i += 256) mx = fmaxf(mx, s[i]);
    mx = block_max(mx, red);

    float sum = 0.f;
    for (int i = tid; i < n; i += 256) sum += __expf(s[i] - mx);
    sum = block_sum(sum, red);
    float inv = 1.0f / sum;

    for (int i = tid; i < n; i += 256)
        p[i] = f32_to_bf16(__expf(s[i] - mx) * inv);
}

// ---------------------------------------------------------------------------
// bf16 GEMM via v_wmma_f32_16x16x32_bf16, double-buffered LDS, async A tiles.
//   C[M,N] = epilogue(alpha * A[M,K] @ B[K,N])
// Block tile: 128 x BN (BN = 64 or 128), BK = 32, 256 threads = 8 waves in a
// 4x2 grid; each wave owns 2 x (BN/32) WMMA 16x16 tiles.
// As[buf][m][k] filled by per-lane global_load_async_to_lds_b128 (16B chunks,
// layout unchanged); Bs[buf][n][k] transposed via register scatter so each
// lane's fragments are contiguous, 16B-aligned ds_load_b128s:
//   A frag: M=l16, K = g*8+i | 16+g*8+(i-8)   B frag: N=l16, K same as A
// M%128==0, N%BN==0, K%32==0 at every call site here.
// ---------------------------------------------------------------------------
template <int BN>
__global__ __launch_bounds__(256) void gemm_bf16_wmma(
    const unsigned short* __restrict__ A, const unsigned short* __restrict__ B,
    int M, int N, int K, int lda, int ldb, float alpha,
    const float* __restrict__ bias,
    const float* __restrict__ res1, const float* __restrict__ res2, int ldres,
    void* __restrict__ Cout, int ldc, int flags) {
    constexpr int BM = 128, BK = 32, PAD = 8, LDW = BK + PAD;
    constexpr int TN  = BN / 32;              // wmma n-tiles per wave
    constexpr int BCH = (BK * BN) / (256 * 8); // B 8-elem chunks per thread

    __shared__ unsigned short As[2][BM][LDW];
    __shared__ unsigned short Bs[2][BN][LDW];

    const int tid  = threadIdx.x;
    const int lane = tid & 31;
    const int wave = tid >> 5;
    const int wm   = wave >> 1;    // 0..3
    const int wn   = wave & 1;     // 0..1
    const int g    = lane >> 4;
    const int l16  = lane & 15;

    const int bm = blockIdx.y * BM;
    const int bn = blockIdx.x * BN;

    v8f acc[2][TN];
    #pragma unroll
    for (int i = 0; i < 2; i++)
        #pragma unroll
        for (int j = 0; j < TN; j++)
            #pragma unroll
            for (int e = 0; e < 8; e++) acc[i][j][e] = 0.f;

    // A: 16 bf16 per thread as two async b128s (row a_row, k chunk a_k0)
    const int a_row = tid & 127;
    const int a_k0  = (tid >> 7) * 16;
    const unsigned short* aptr = A + (size_t)(bm + a_row) * lda + a_k0;

    // B: BCH chunks of 8 contiguous bf16 per thread
    int bk_[BCH], bn_[BCH];
    const unsigned short* bptr[BCH];
    #pragma unroll
    for (int c = 0; c < BCH; c++) {
        const int gid = c * 256 + tid;
        bk_[c] = gid / (BN / 8);
        bn_[c] = (gid % (BN / 8)) * 8;
        bptr[c] = B + (size_t)bk_[c] * ldb + bn + bn_[c];
    }

    u16x8 breg[BCH];
    auto issueA = [&](int buf) {
        async_load_b128(aptr,     &As[buf][a_row][a_k0]);
        async_load_b128(aptr + 8, &As[buf][a_row][a_k0 + 8]);
        aptr += BK;
    };
    auto loadB = [&]() {
        #pragma unroll
        for (int c = 0; c < BCH; c++) {
            breg[c] = *(const u16x8*)bptr[c];
            __builtin_prefetch(bptr[c] + (size_t)BK * ldb, 0, 1);
            bptr[c] += (size_t)BK * ldb;
        }
    };
    auto scatterB = [&](int buf) {
        #pragma unroll
        for (int c = 0; c < BCH; c++)
            #pragma unroll
            for (int j = 0; j < 8; j++)
                Bs[buf][bn_[c] + j][bk_[c]] = breg[c][j];
    };

    // prologue: tile 0 in flight
    issueA(0);
    loadB();
    scatterB(0);

    const int nk = K / BK;
    for (int i = 0; i < nk; i++) {
        const int cur = i & 1;
        wait_async0();        // async A writes of tile i complete
        __syncthreads();      // + Bs scatter of tile i visible to all waves

        if (i + 1 < nk) {     // overlap tile i+1 fetch with tile i compute
            issueA(cur ^ 1);
            loadB();
        }

        v16bf af[2];
        #pragma unroll
        for (int t = 0; t < 2; t++) {
            const unsigned short* pa = &As[cur][wm * 32 + t * 16 + l16][g * 8];
            v8bf alo = *(const v8bf*)pa;
            v8bf ahi = *(const v8bf*)(pa + 16);
            #pragma unroll
            for (int e = 0; e < 8; e++) { af[t][e] = alo[e]; af[t][8 + e] = ahi[e]; }
        }
        v16bf bfr[TN];
        #pragma unroll
        for (int t = 0; t < TN; t++) {
            const unsigned short* pb = &Bs[cur][wn * (BN / 2) + t * 16 + l16][g * 8];
            v8bf blo = *(const v8bf*)pb;
            v8bf bhi = *(const v8bf*)(pb + 16);
            #pragma unroll
            for (int e = 0; e < 8; e++) { bfr[t][e] = blo[e]; bfr[t][8 + e] = bhi[e]; }
        }

        #pragma unroll
        for (int tm = 0; tm < 2; tm++)
            #pragma unroll
            for (int tn = 0; tn < TN; tn++)
                acc[tm][tn] = __builtin_amdgcn_wmma_f32_16x16x32_bf16(
                    false, af[tm], false, bfr[tn], (short)0, acc[tm][tn],
                    false, false);

        if (i + 1 < nk) scatterB(cur ^ 1);   // other buffer: no barrier needed
    }

    // epilogue; C f32 layout: VGPR r -> M = r + 8*g, N = l16
    #pragma unroll
    for (int tm = 0; tm < 2; tm++) {
        #pragma unroll
        for (int tn = 0; tn < TN; tn++) {
            const int mb = bm + wm * 32 + tm * 16 + g * 8;
            const int nn = bn + wn * (BN / 2) + tn * 16 + l16;
            const float bv_ = bias ? bias[nn] : 0.f;
            #pragma unroll
            for (int r = 0; r < 8; r++) {
                const int m = mb + r;
                float v = acc[tm][tn][r] * alpha + bv_;
                if (res1) v += res1[(size_t)m * ldres + nn];
                if (res2) v += res2[(size_t)m * ldres + nn];
                if (flags & GF_RELU) v = fmaxf(v, 0.f);
                const size_t ci = (flags & GF_TRANS) ? ((size_t)nn * ldc + m)
                                                     : ((size_t)m * ldc + nn);
                if (flags & GF_BF16) ((unsigned short*)Cout)[ci] = f32_to_bf16(v);
                else                 ((float*)Cout)[ci] = v;
            }
        }
    }
}

// ---------------------------------------------------------------------------
extern "C" void kernel_launch(void* const* d_in, const int* in_sizes, int n_in,
                              void* d_out, int out_size, void* d_ws, size_t ws_size,
                              hipStream_t stream) {
    const float* x   = (const float*)d_in[0];
    const float* Wq  = (const float*)d_in[1];
    const float* Wk  = (const float*)d_in[2];
    const float* Wv  = (const float*)d_in[3];
    const float* Wo  = (const float*)d_in[4];
    const float* bo  = (const float*)d_in[5];
    const float* W1  = (const float*)d_in[6];
    const float* b1  = (const float*)d_in[7];
    const float* W2  = (const float*)d_in[8];
    const float* b2  = (const float*)d_in[9];
    const float* g1  = (const float*)d_in[10];
    const float* be1 = (const float*)d_in[11];
    const float* g2  = (const float*)d_in[12];
    const float* be2 = (const float*)d_in[13];
    float* out = (float*)d_out;

    // ---- workspace carve-up (256B aligned) ----
    size_t off = 0;
    auto wsalloc = [&](size_t bytes) -> char* {
        char* p = (char*)d_ws + off;
        off += (bytes + 255) & ~(size_t)255;
        return p;
    };
    unsigned short* h1   = (unsigned short*)wsalloc((size_t)Bb * Tt * Ee * 2);
    unsigned short* h2   = (unsigned short*)wsalloc((size_t)Bb * Tt * Ee * 2);
    unsigned short* Wq_b = (unsigned short*)wsalloc((size_t)Hh * Ee * DHd * 2);
    unsigned short* Wk_b = (unsigned short*)wsalloc((size_t)Hh * Ee * DHd * 2);
    unsigned short* Wv_b = (unsigned short*)wsalloc((size_t)Hh * Ee * DHd * 2);
    unsigned short* Wo_b = (unsigned short*)wsalloc((size_t)Ee * Ee * 2);
    unsigned short* W1_b = (unsigned short*)wsalloc((size_t)Ee * FFf * 2);
    unsigned short* W2_b = (unsigned short*)wsalloc((size_t)FFf * Ee * 2);
    unsigned short* q_b  = (unsigned short*)wsalloc((size_t)Bb * Hh * Tt * DHd * 2);
    unsigned short* kT_b = (unsigned short*)wsalloc((size_t)Bb * Hh * DHd * Tt * 2);
    unsigned short* v_b  = (unsigned short*)wsalloc((size_t)Bb * Hh * Tt * DHd * 2);
    unsigned short* o_b  = (unsigned short*)wsalloc((size_t)Bb * Tt * Ee * 2);
    unsigned short* ff1b = (unsigned short*)wsalloc((size_t)Bb * Tt * FFf * 2);
    float*          S_ws = (float*)wsalloc((size_t)Tt * Tt * 4);
    unsigned short* P_b  = (unsigned short*)wsalloc((size_t)Tt * Tt * 2);
    float*          attn = (float*)wsalloc((size_t)Bb * Tt * Ee * 4);

    auto gemm = [&](const unsigned short* A, const unsigned short* B,
                    int M, int N, int K, int lda, int ldb, float alpha,
                    const float* bias, const float* r1, const float* r2, int ldres,
                    void* C, int ldc, int flags) {
        if (N % 128 == 0) {
            dim3 grid(N / 128, M / 128);
            gemm_bf16_wmma<128><<<grid, dim3(256), 0, stream>>>(
                A, B, M, N, K, lda, ldb, alpha, bias, r1, r2, ldres, C, ldc, flags);
        } else {
            dim3 grid(N / 64, M / 128);
            gemm_bf16_wmma<64><<<grid, dim3(256), 0, stream>>>(
                A, B, M, N, K, lda, ldb, alpha, bias, r1, r2, ldres, C, ldc, flags);
        }
    };
    auto cvt = [&](const float* in, unsigned short* o, int n) {
        cvt_f32_bf16_kernel<<<(n + 255) / 256, 256, 0, stream>>>(in, o, n);
    };

    // 1) LayerNorms (bf16 activations)
    ln_bf16_kernel<<<Bb * Tt, 256, 0, stream>>>(x, g1, be1, h1, Ee);
    ln_bf16_kernel<<<Bb * Tt, 256, 0, stream>>>(x, g2, be2, h2, Ee);

    // 2) weight conversion fp32 -> bf16
    cvt(Wq, Wq_b, Hh * Ee * DHd);
    cvt(Wk, Wk_b, Hh * Ee * DHd);
    cvt(Wv, Wv_b, Hh * Ee * DHd);
    cvt(Wo, Wo_b, Ee * Ee);
    cvt(W1, W1_b, Ee * FFf);
    cvt(W2, W2_b, FFf * Ee);

    // 3) Q, K^T, V projections (per batch, per head)
    for (int b = 0; b < Bb; b++) {
        const unsigned short* Ab = h1 + (size_t)b * Tt * Ee;
        for (int h = 0; h < Hh; h++) {
            const size_t bh = (size_t)(b * Hh + h);
            gemm(Ab, Wq_b + (size_t)h * Ee * DHd, Tt, DHd, Ee, Ee, DHd, 1.f,
                 nullptr, nullptr, nullptr, 0,
                 q_b + bh * Tt * DHd, DHd, GF_BF16);
            gemm(Ab, Wk_b + (size_t)h * Ee * DHd, Tt, DHd, Ee, Ee, DHd, 1.f,
                 nullptr, nullptr, nullptr, 0,
                 kT_b + bh * DHd * Tt, Tt, GF_BF16 | GF_TRANS);   // store K^T
            gemm(Ab, Wv_b + (size_t)h * Ee * DHd, Tt, DHd, Ee, Ee, DHd, 1.f,
                 nullptr, nullptr, nullptr, 0,
                 v_b + bh * Tt * DHd, DHd, GF_BF16);
        }
    }

    // 4) attention per (b,h): S = 0.125 * Q K^T ; P = softmax(S) ; O = P V
    for (int bh = 0; bh < Bb * Hh; bh++) {
        const unsigned short* qp  = q_b  + (size_t)bh * Tt * DHd;
        const unsigned short* ktp = kT_b + (size_t)bh * DHd * Tt;
        const unsigned short* vp  = v_b  + (size_t)bh * Tt * DHd;
        gemm(qp, ktp, Tt, Tt, DHd, DHd, Tt, 0.125f,
             nullptr, nullptr, nullptr, 0, S_ws, Tt, 0);
        softmax_bf16_kernel<<<Tt, 256, 0, stream>>>(S_ws, P_b, Tt);
        const int b = bh / Hh, h = bh % Hh;
        gemm(P_b, vp, Tt, DHd, Tt, Tt, DHd, 1.f,
             nullptr, nullptr, nullptr, 0,
             o_b + (size_t)b * Tt * Ee + (size_t)h * DHd, Ee, GF_BF16);
    }

    // 5) output projection: attn = O @ Wo + bo   (fp32)
    gemm(o_b, Wo_b, Bb * Tt, Ee, Ee, Ee, Ee, 1.f,
         bo, nullptr, nullptr, 0, attn, Ee, 0);

    // 6) FFN: ff1 = relu(h2 @ W1 + b1) (bf16)
    gemm(h2, W1_b, Bb * Tt, FFf, Ee, Ee, FFf, 1.f,
         b1, nullptr, nullptr, 0, ff1b, FFf, GF_RELU | GF_BF16);

    // 7) out = ff1 @ W2 + b2 + x + attn  (fused residuals, fp32)
    gemm(ff1b, W2_b, Bb * Tt, Ee, FFf, FFf, Ee, 1.f,
         b2, x, attn, Ee, out, Ee, 0);
}
